// BinaryTreeLSTM_73160472920486
// MI455X (gfx1250) — matrix operand order repaired and checked by reading, not compiled, not run on hardware
//
#include <hip/hip_runtime.h>

#define HIDDEN 256
#define TILE_M 32          // workgroup M tile = 2 WMMA subtiles of 16
#define NLEAVES 32768

typedef __attribute__((ext_vector_type(16))) __bf16       v16bf;
typedef __attribute__((ext_vector_type(8)))  float        v8f;
typedef __attribute__((ext_vector_type(4)))  unsigned int u32x4;
typedef __attribute__((ext_vector_type(4)))  float        f32x4;

union FragAB { u32x4 u[2]; v16bf v; };

__device__ __forceinline__ float sigm(float x) { return 1.0f / (1.0f + __expf(-x)); }

// ---------------------------------------------------------------------------
// f32 -> bf16 weight convert (B column-major flat order == W's own [g][e][k]).
// ---------------------------------------------------------------------------
__global__ __launch_bounds__(256)
void convert_weights(const float* __restrict__ Wx, const float* __restrict__ Wl,
                     const float* __restrict__ Wr,
                     __bf16* __restrict__ dWx, __bf16* __restrict__ dWl,
                     __bf16* __restrict__ dWr) {
  int i = blockIdx.x * blockDim.x + threadIdx.x;   // 0 .. 4*256*256-1
  dWx[i] = (__bf16)Wx[i];
  dWl[i] = (__bf16)Wl[i];
  dWr[i] = (__bf16)Wr[i];
}

// A 16x32 bf16 fragment from an LDS 16x256 row-major tile (rowBase selects
// the M-subtile): elems 0..7 -> K = kb+8*half+{0..7}, elems 8..15 -> +16.
__device__ __forceinline__ v16bf load_a_frag(const __bf16* lds, int lane, int kb) {
  const int m = lane & 15, half = lane >> 4;
  const u32x4* p = reinterpret_cast<const u32x4*>(lds + m * HIDDEN + kb + 8 * half);
  FragAB a;
  a.u[0] = p[0];
  a.u[1] = p[2];   // +32 bytes
  return a.v;
}

// B 32x16 bf16 fragment: lane(half,n) holds K = kb + 8*half + {0..7} and
// +16 of its column; column-major weights => two b128 loads.
__device__ __forceinline__ v16bf load_b_frag(const __bf16* W, int col, int lane, int kb) {
  const int half = lane >> 4;
  const u32x4* p = reinterpret_cast<const u32x4*>(W + (size_t)col * HIDDEN + kb + 8 * half);
  FragAB b;
  b.u[0] = p[0];
  b.u[1] = p[2];
  return b.v;
}

// ---------------------------------------------------------------------------
// Leaf level: gather emb[tokens] -> LDS bf16 (32 rows), z = x*Wx + b,
// c = i*u, h = o*tanh(c). Each wave: 4 gates x 2 hidden tiles x 2 M-subtiles.
// ---------------------------------------------------------------------------
__global__ __launch_bounds__(256)
void leaf_kernel(const int* __restrict__ tokens, const float* __restrict__ emb,
                 const __bf16* __restrict__ Wx, const float* __restrict__ bias,
                 float* __restrict__ out, __bf16* __restrict__ hbf,
                 float* __restrict__ cbuf) {
  __shared__ __align__(16) __bf16 xs[TILE_M * HIDDEN];
  const int tid = threadIdx.x;
  const int tileBase = blockIdx.x * TILE_M;

  // Stage 32 embedding rows (f32 -> bf16): 32 rows * 64 float4 chunks.
  for (int i = tid; i < TILE_M * 64; i += 256) {
    const int row = i >> 6, ch = i & 63;
    const int tok = tokens[tileBase + row];
    const f32x4 v = reinterpret_cast<const f32x4*>(emb + (size_t)tok * HIDDEN)[ch];
    __bf16* dst = &xs[row * HIDDEN + ch * 4];
    dst[0] = (__bf16)v.x; dst[1] = (__bf16)v.y;
    dst[2] = (__bf16)v.z; dst[3] = (__bf16)v.w;
  }
  __syncthreads();

  const int lane = tid & 31, w = tid >> 5;
  const int n = lane & 15, half = lane >> 4;

  const v8f vzero = {0.f, 0.f, 0.f, 0.f, 0.f, 0.f, 0.f, 0.f};
  v8f acc[4][2][2];                       // [gate][hidden j][M-subtile]
#pragma unroll
  for (int g = 0; g < 4; ++g)
#pragma unroll
    for (int j = 0; j < 2; ++j) { acc[g][j][0] = vzero; acc[g][j][1] = vzero; }

#pragma unroll
  for (int ks = 0; ks < 8; ++ks) {        // K = 256 in steps of 32
    const int kb = ks * 32;
    const v16bf a0 = load_a_frag(xs, lane, kb);
    const v16bf a1 = load_a_frag(xs + 16 * HIDDEN, lane, kb);
#pragma unroll
    for (int g = 0; g < 4; ++g)
#pragma unroll
      for (int j = 0; j < 2; ++j) {
        const int col = g * 256 + (2 * w + j) * 16 + n;
        const v16bf bfrag = load_b_frag(Wx, col, lane, kb);   // feeds 2 WMMAs
        acc[g][j][0] = __builtin_amdgcn_wmma_f32_16x16x32_bf16(
            false, a0, false, bfrag, (short)0, acc[g][j][0], false, false);
        acc[g][j][1] = __builtin_amdgcn_wmma_f32_16x16x32_bf16(
            false, a1, false, bfrag, (short)0, acc[g][j][1], false, false);
      }
  }

  // Gate math in registers (C layout: M = r + 8*half, N = lane&15).
#pragma unroll
  for (int j = 0; j < 2; ++j) {
    const int kcol = (2 * w + j) * 16 + n;
    const float b0 = bias[0 * 256 + kcol];
    const float b2 = bias[2 * 256 + kcol];
    const float b3 = bias[3 * 256 + kcol];
#pragma unroll
    for (int s = 0; s < 2; ++s)
#pragma unroll
      for (int r = 0; r < 8; ++r) {
        const int m = s * 16 + r + 8 * half;
        const float ig = sigm(acc[0][j][s][r] + b0);
        const float og = sigm(acc[2][j][s][r] + b2);
        const float ug = tanhf(acc[3][j][s][r] + b3);
        const float c = ig * ug;
        const float h = og * tanhf(c);
        const size_t idx = (size_t)(tileBase + m) * HIDDEN + kcol;
        out[idx]  = h;
        hbf[idx]  = (__bf16)h;
        cbuf[idx] = c;
      }
  }
}

// ---------------------------------------------------------------------------
// Internal level: z = lh*Wl + rh*Wr + b; c = i*u + f*(lc+rc); h = o*tanh(c).
// ---------------------------------------------------------------------------
__global__ __launch_bounds__(256)
void node_kernel(const __bf16* __restrict__ hbf, const float* __restrict__ cbuf,
                 const __bf16* __restrict__ Wl, const __bf16* __restrict__ Wr,
                 const float* __restrict__ bias, float* __restrict__ out,
                 int nNodes, int levelBase, int prevBase) {
  __shared__ __align__(16) __bf16 lhs[TILE_M * HIDDEN];
  __shared__ __align__(16) __bf16 rhs[TILE_M * HIDDEN];
  const int tid = threadIdx.x;
  const int tileBase = blockIdx.x * TILE_M;

  // Stage 32 (lh, rh) rows as b128 copies; zero-pad past nNodes.
  for (int i = tid; i < TILE_M * 32; i += 256) {   // 32 rows * 32 uint4/row
    const int row = i >> 5, ch = i & 31;
    const int m = tileBase + row;
    u32x4 lv = {0u, 0u, 0u, 0u}, rv = {0u, 0u, 0u, 0u};
    if (m < nNodes) {
      const size_t base = (size_t)(prevBase + 2 * m) * HIDDEN;
      lv = reinterpret_cast<const u32x4*>(hbf + base)[ch];
      rv = reinterpret_cast<const u32x4*>(hbf + base + HIDDEN)[ch];
    }
    reinterpret_cast<u32x4*>(lhs)[i] = lv;
    reinterpret_cast<u32x4*>(rhs)[i] = rv;
  }
  __syncthreads();

  const int lane = tid & 31, w = tid >> 5;
  const int n = lane & 15, half = lane >> 4;

  const v8f vzero = {0.f, 0.f, 0.f, 0.f, 0.f, 0.f, 0.f, 0.f};
  v8f acc[4][2][2];
#pragma unroll
  for (int g = 0; g < 4; ++g)
#pragma unroll
    for (int j = 0; j < 2; ++j) { acc[g][j][0] = vzero; acc[g][j][1] = vzero; }

#pragma unroll
  for (int ks = 0; ks < 8; ++ks) {
    const int kb = ks * 32;
    const v16bf aL0 = load_a_frag(lhs, lane, kb);
    const v16bf aL1 = load_a_frag(lhs + 16 * HIDDEN, lane, kb);
    const v16bf aR0 = load_a_frag(rhs, lane, kb);
    const v16bf aR1 = load_a_frag(rhs + 16 * HIDDEN, lane, kb);
#pragma unroll
    for (int g = 0; g < 4; ++g)
#pragma unroll
      for (int j = 0; j < 2; ++j) {
        const int col = g * 256 + (2 * w + j) * 16 + n;
        const v16bf bL = load_b_frag(Wl, col, lane, kb);
        acc[g][j][0] = __builtin_amdgcn_wmma_f32_16x16x32_bf16(
            false, aL0, false, bL, (short)0, acc[g][j][0], false, false);
        acc[g][j][1] = __builtin_amdgcn_wmma_f32_16x16x32_bf16(
            false, aL1, false, bL, (short)0, acc[g][j][1], false, false);
        const v16bf bR = load_b_frag(Wr, col, lane, kb);
        acc[g][j][0] = __builtin_amdgcn_wmma_f32_16x16x32_bf16(
            false, aR0, false, bR, (short)0, acc[g][j][0], false, false);
        acc[g][j][1] = __builtin_amdgcn_wmma_f32_16x16x32_bf16(
            false, aR1, false, bR, (short)0, acc[g][j][1], false, false);
      }
  }

#pragma unroll
  for (int j = 0; j < 2; ++j) {
    const int kcol = (2 * w + j) * 16 + n;
    const float b0 = bias[0 * 256 + kcol];
    const float b1 = bias[1 * 256 + kcol];
    const float b2 = bias[2 * 256 + kcol];
    const float b3 = bias[3 * 256 + kcol];
#pragma unroll
    for (int s = 0; s < 2; ++s)
#pragma unroll
      for (int r = 0; r < 8; ++r) {
        const int m = tileBase + s * 16 + r + 8 * half;
        if (m < nNodes) {
          const float ig = sigm(acc[0][j][s][r] + b0);
          const float fg = sigm(acc[1][j][s][r] + b1);
          const float og = sigm(acc[2][j][s][r] + b2);
          const float ug = tanhf(acc[3][j][s][r] + b3);
          const size_t cidx = (size_t)(prevBase + 2 * m) * HIDDEN + kcol;
          const float lc = cbuf[cidx];
          const float rc = cbuf[cidx + HIDDEN];
          const float c = ig * ug + fg * (lc + rc);
          const float h = og * tanhf(c);
          const size_t idx = (size_t)(levelBase + m) * HIDDEN + kcol;
          out[idx] = h;
          const_cast<__bf16*>(hbf)[idx] = (__bf16)h;
          const_cast<float*>(cbuf)[idx] = c;
        }
      }
  }
}

// ---------------------------------------------------------------------------
// Launch: convert weights, leaf level, then 15 halving levels in stream order.
// ---------------------------------------------------------------------------
extern "C" void kernel_launch(void* const* d_in, const int* in_sizes, int n_in,
                              void* d_out, int out_size, void* d_ws, size_t ws_size,
                              hipStream_t stream) {
  const int*   tokens = (const int*)d_in[0];
  const float* emb    = (const float*)d_in[1];
  const float* Wx     = (const float*)d_in[2];
  const float* Wl     = (const float*)d_in[3];
  const float* Wr     = (const float*)d_in[4];
  const float* bias   = (const float*)d_in[5];
  float* out = (float*)d_out;

  char* ws = (char*)d_ws;
  __bf16* dWx = (__bf16*)(ws);                    //  512 KB
  __bf16* dWl = (__bf16*)(ws + (512u << 10));     //  512 KB
  __bf16* dWr = (__bf16*)(ws + (1024u << 10));    //  512 KB
  __bf16* hbf = (__bf16*)(ws + (1536u << 10));    // 65536*256*2 = 32 MB
  float*  cb  = (float*) (ws + (1536u << 10) + ((size_t)65536 * HIDDEN * 2)); // 64 MB

  convert_weights<<<(4 * 256 * 256) / 256, 256, 0, stream>>>(Wx, Wl, Wr, dWx, dWl, dWr);
  leaf_kernel<<<NLEAVES / TILE_M, 256, 0, stream>>>(tokens, emb, dWx, bias, out, hbf, cb);

  int levelBase = 0;
  int nPrev = NLEAVES;
  while (nPrev > 1) {
    const int prevBase = levelBase;
    levelBase += nPrev;
    const int n = nPrev >> 1;
    const int blocks = (n + TILE_M - 1) / TILE_M;
    node_kernel<<<blocks, 256, 0, stream>>>(hbf, cb, dWl, dWr, bias, out,
                                            n, levelBase, prevBase);
    nPrev = n;
  }
}
